// HomoNeighborhoodAttention_24232205484252
// MI455X (gfx1250) — matrix operand order repaired and checked by reading, not compiled, not run on hardware
//
#include <hip/hip_runtime.h>

#define N_NODES 50000
#define N_EDGES 800000
#define COMB    144
#define KPAD    160

typedef __attribute__((ext_vector_type(16))) _Float16 v16h;
typedef __attribute__((ext_vector_type(8)))  _Float16 v8h;
typedef __attribute__((ext_vector_type(8)))  float    v8f;

__device__ __forceinline__ float reluf(float x){ return x > 0.f ? x : 0.f; }

__device__ __forceinline__ v8f wmma16(v16h a, v16h b, v8f c){
  // D = A(16x32 f16) x B(32x16 f16) + C(f32)
  return __builtin_amdgcn_wmma_f32_16x16x32_f16(false, a, false, b, (short)0, c, false, false);
}

// A-fragment (16x32 f16) from LDS row-major tile, per ISA 7.12.2:
// lane m=lane&15, half=lane>>4; VGPR v holds K = 16*(v>>2) + 8*half + 2*(v&3) + {0,1}
__device__ __forceinline__ v16h lds_load_a(const _Float16* row, int ks, int hlf){
  v8h lo = *(const v8h*)(row + ks + hlf*8);        // K = ks + 8*half .. +7
  v8h hi = *(const v8h*)(row + ks + 16 + hlf*8);   // K = ks+16+8*half .. +7
  v16h r;
#pragma unroll
  for (int i=0;i<8;++i){ r[i]=lo[i]; r[i+8]=hi[i]; }
  return r;
}

// B-fragment (32x16 f16) from LDS weights W[k][n] (row stride 64):
// lane n=lane&15, half=lane>>4; elements hold K = k0 + 8*half + i and k0 + 16 + 8*half + i (same as A)
__device__ __forceinline__ v16h lds_load_b(const _Float16* W, int k0, int n0, int lane){
  const int hb = lane >> 4, nb = n0 + (lane & 15);
  v16h r;
#pragma unroll
  for (int i=0;i<8;++i){ r[i]=W[(k0+8*hb+i)*64+nb]; r[i+8]=W[(k0+16+8*hb+i)*64+nb]; }
  return r;
}

// order-preserving float<->uint encoding for atomicMax-based segment max
__device__ __forceinline__ unsigned enc_f32(float f){
  unsigned u = __float_as_uint(f);
  return (u & 0x80000000u) ? ~u : (u | 0x80000000u);
}
__device__ __forceinline__ float dec_f32(unsigned e){
  return (e & 0x80000000u) ? __uint_as_float(e & 0x7FFFFFFFu) : __uint_as_float(~e);
}

// layers 1 and 2 of a residual stack: h = relu(h + h@W1 + b1); h = h + h@W2 + b2
// c[] holds h (f32 frags), hT holds h (f16) as A-source; both updated.
__device__ __forceinline__ void residual_tail(
    v8f c[4], _Float16* hT, int m, int hlf, int lane,
    const _Float16* W1, const float* b1, const _Float16* W2, const float* b2)
{
  const int n = lane & 15;
  {
    v16h a0 = lds_load_a(hT + m*64, 0, hlf);
    v16h a1 = lds_load_a(hT + m*64, 32, hlf);
#pragma unroll
    for (int t=0;t<4;++t){
      c[t] = wmma16(a0, lds_load_b(W1, 0,  t*16, lane), c[t]);
      c[t] = wmma16(a1, lds_load_b(W1, 32, t*16, lane), c[t]);
#pragma unroll
      for (int v=0;v<8;++v) c[t][v] = reluf(c[t][v] + b1[t*16+n]);
    }
#pragma unroll
    for (int t=0;t<4;++t)
#pragma unroll
      for (int v=0;v<8;++v) hT[(v+8*hlf)*64 + t*16 + n] = (_Float16)c[t][v];
  }
  {
    v16h a0 = lds_load_a(hT + m*64, 0, hlf);
    v16h a1 = lds_load_a(hT + m*64, 32, hlf);
#pragma unroll
    for (int t=0;t<4;++t){
      c[t] = wmma16(a0, lds_load_b(W2, 0,  t*16, lane), c[t]);
      c[t] = wmma16(a1, lds_load_b(W2, 32, t*16, lane), c[t]);
#pragma unroll
      for (int v=0;v<8;++v) c[t][v] = c[t][v] + b2[t*16+n];
    }
  }
}

// ---------------- Kernel 0: zero segment-reduction buffers ----------------
__global__ __launch_bounds__(256) void init_ws_kernel(
    unsigned* __restrict__ mBuf, float* __restrict__ denomBuf, float* __restrict__ accBuf)
{
  int g = blockIdx.x*256 + threadIdx.x;
  if (g < N_NODES*8){ mBuf[g] = 0u; denomBuf[g] = 0.f; }   // enc(-NaN/min) = 0
  if (g < N_NODES*64) accBuf[g] = 0.f;
}

// ---------------- Kernel 1: fused edge k/v stacks + scores (WMMA) ----------------
__global__ __launch_bounds__(256) void edge_kv_kernel(
    const float* __restrict__ x, const float* __restrict__ ea,
    const int* __restrict__ ei, const float* __restrict__ q,
    const float* kW0, const float* kb0, const float* kW1, const float* kb1,
    const float* kW2, const float* kb2,
    const float* vW0, const float* vb0, const float* vW1, const float* vb1,
    const float* vW2, const float* vb2,
    float* __restrict__ sBuf, _Float16* __restrict__ vBuf, unsigned* __restrict__ mBuf)
{
  __shared__ __align__(32) _Float16 sW0[2][KPAD*64];  // K padded 144->160 with zeros
  __shared__ __align__(32) _Float16 sW1[2][64*64];
  __shared__ __align__(32) _Float16 sW2[2][64*64];
  __shared__ float sB0[2][64], sB1[2][64], sB2[2][64];
  __shared__ float sQ[64];
  __shared__ __align__(32) _Float16 sH[8][16*64];     // per-wave staging tile

  const int tid = threadIdx.x;
  for (int i=tid;i<COMB*64;i+=256){ sW0[0][i]=(_Float16)kW0[i]; sW0[1][i]=(_Float16)vW0[i]; }
  for (int i=tid;i<(KPAD-COMB)*64;i+=256){
    sW0[0][COMB*64+i]=(_Float16)0.f; sW0[1][COMB*64+i]=(_Float16)0.f;
  }
  for (int i=tid;i<64*64;i+=256){
    sW1[0][i]=(_Float16)kW1[i]; sW1[1][i]=(_Float16)vW1[i];
    sW2[0][i]=(_Float16)kW2[i]; sW2[1][i]=(_Float16)vW2[i];
  }
  if (tid<64){
    sB0[0][tid]=kb0[tid]; sB0[1][tid]=vb0[tid];
    sB1[0][tid]=kb1[tid]; sB1[1][tid]=vb1[tid];
    sB2[0][tid]=kb2[tid]; sB2[1][tid]=vb2[tid];
    sQ[tid]=q[tid];
  }
  __syncthreads();

  const int wave = tid>>5, lane = tid&31;
  const int m = lane&15, hlf = lane>>4, n = lane&15;
  const int eBase = blockIdx.x*128 + wave*16;   // E divisible by 128
  const int e = eBase + m;
  const int srcN = ei[e];
  const int dstN = ei[N_EDGES + e];

  // Build feat A-fragments (16 edges x 160) straight from global gathers.
  // feat = [x[src](64) | x[dst](64) | edge_attr(16) | zero-pad(16)]
  v16h af[5];
#pragma unroll
  for (int kt=0; kt<5; ++kt){
#pragma unroll
    for (int v=0; v<8; ++v){
      int kk = kt*32 + (v>>2)*16 + hlf*8 + (v&3)*2;  // even; pairs never straddle regions
      float f0, f1;
      if (kk < 64)        { f0 = x[srcN*64+kk];     f1 = x[srcN*64+kk+1]; }
      else if (kk < 128)  { f0 = x[dstN*64+kk-64];  f1 = x[dstN*64+kk-63]; }
      else if (kk < COMB) { f0 = ea[e*16+kk-128];   f1 = ea[e*16+kk-127]; }
      else                { f0 = 0.f; f1 = 0.f; }
      af[kt][2*v]   = (_Float16)f0;
      af[kt][2*v+1] = (_Float16)f1;
    }
  }

  _Float16* hT = sH[wave];

#pragma unroll 1
  for (int s=0; s<2; ++s){                      // s=0: k stack, s=1: v stack
    const _Float16* W0s = sW0[s]; const _Float16* W1s = sW1[s]; const _Float16* W2s = sW2[s];
    const float* b0s = sB0[s]; const float* b1s = sB1[s]; const float* b2s = sB2[s];

    v8f c[4];
#pragma unroll
    for (int t=0;t<4;++t){
#pragma unroll
      for (int v=0;v<8;++v) c[t][v]=0.f;
#pragma unroll
      for (int kt=0;kt<5;++kt)
        c[t] = wmma16(af[kt], lds_load_b(W0s, kt*32, t*16, lane), c[t]);
#pragma unroll
      for (int v=0;v<8;++v) c[t][v] = reluf(c[t][v] + b0s[t*16+n]);
    }
#pragma unroll
    for (int t=0;t<4;++t)
#pragma unroll
      for (int v=0;v<8;++v) hT[(v+8*hlf)*64 + t*16 + n] = (_Float16)c[t][v];

    residual_tail(c, hT, m, hlf, lane, W1s, b1s, W2s, b2s);

    if (s == 0){
      // stage k (f16) then per-(edge,head) dot with q; encoded atomicMax for segment max
#pragma unroll
      for (int t=0;t<4;++t)
#pragma unroll
        for (int v=0;v<8;++v) hT[(v+8*hlf)*64 + t*16 + n] = (_Float16)c[t][v];
      const float inv = 0.3535533905932738f;    // 1/sqrt(HEAD_DIM=8)
#pragma unroll
      for (int i=0;i<4;++i){
        int idx = lane*4 + i;                   // 16 edges x 8 heads = 128 per wave
        int el = idx>>3, hh = idx&7;
        float sc = 0.f;
#pragma unroll
        for (int j=0;j<8;++j) sc += sQ[hh*8+j] * (float)hT[el*64 + hh*8 + j];
        sc *= inv;
        int eg = eBase + el;
        sBuf[eg*8 + hh] = sc;
        int dn = ei[N_EDGES + eg];
        atomicMax(&mBuf[dn*8 + hh], enc_f32(sc));
      }
    } else {
      // write v (f16) straight from D fragments
#pragma unroll
      for (int t=0;t<4;++t)
#pragma unroll
        for (int v=0;v<8;++v)
          vBuf[(size_t)(eBase + v + 8*hlf)*64 + t*16 + n] = (_Float16)c[t][v];
    }
  }
}

// ---------------- Kernel 2: softmax numerators + fused scatter of ex*v ----------------
__global__ __launch_bounds__(256) void edge_softmax_scatter(
    const int* __restrict__ ei, const float* __restrict__ sBuf,
    const _Float16* __restrict__ vBuf, const unsigned* __restrict__ mBuf,
    float* __restrict__ denomBuf, float* __restrict__ accBuf)
{
  size_t g = (size_t)blockIdx.x*256 + threadIdx.x;   // one thread per (edge, head)
  int e  = (int)(g>>3), hh = (int)(g&7);
  int d  = ei[N_EDGES + e];
  float s  = sBuf[g];
  float mv = dec_f32(mBuf[d*8+hh]);
  float ex = __expf(s - mv);
  atomicAdd(&denomBuf[d*8+hh], ex);
  const _Float16* vp = vBuf + (size_t)e*64 + hh*8;
#pragma unroll
  for (int j=0;j<8;++j) atomicAdd(&accBuf[d*64 + hh*8 + j], ex * (float)vp[j]);
}

// ---------------- Kernel 3: node output stack (WMMA) + skip ----------------
__global__ __launch_bounds__(256) void node_out_kernel(
    const float* __restrict__ x,
    const float* oW0, const float* ob0, const float* oW1, const float* ob1,
    const float* oW2, const float* ob2,
    const float* __restrict__ denomBuf, const float* __restrict__ accBuf,
    float* __restrict__ out)
{
  __shared__ __align__(32) _Float16 sW0[64*64];
  __shared__ __align__(32) _Float16 sW1[64*64];
  __shared__ __align__(32) _Float16 sW2[64*64];
  __shared__ float sB0[64], sB1[64], sB2[64];
  __shared__ __align__(32) _Float16 sH[8][16*64];

  const int tid = threadIdx.x;
  for (int i=tid;i<64*64;i+=256){
    sW0[i]=(_Float16)oW0[i]; sW1[i]=(_Float16)oW1[i]; sW2[i]=(_Float16)oW2[i];
  }
  if (tid<64){ sB0[tid]=ob0[tid]; sB1[tid]=ob1[tid]; sB2[tid]=ob2[tid]; }
  __syncthreads();

  const int wave=tid>>5, lane=tid&31;
  const int m=lane&15, hlf=lane>>4, n=lane&15;
  const int base = blockIdx.x*128 + wave*16;
  _Float16* hT = sH[wave];

  // h0 = relu(aggr) where aggr = acc/denom (0 for isolated nodes); clamp tail loads
  int nd = base + m;
  int ndc = nd < N_NODES ? nd : (N_NODES-1);
#pragma unroll
  for (int j=0;j<32;++j){
    int col = hlf*32 + j;                     // lane pair (m, m+16) splits the row
    float a   = accBuf[ndc*64 + col];
    float den = denomBuf[ndc*8 + (col>>3)];
    float h   = (den > 0.f) ? (a/den) : 0.f;
    hT[m*64+col] = (_Float16)reluf(h);
  }

  v8f c[4];
  {
    v16h a0 = lds_load_a(hT + m*64, 0, hlf);
    v16h a1 = lds_load_a(hT + m*64, 32, hlf);
#pragma unroll
    for (int t=0;t<4;++t){
#pragma unroll
      for (int v=0;v<8;++v) c[t][v]=0.f;
      c[t]=wmma16(a0, lds_load_b(sW0, 0,  t*16, lane), c[t]);
      c[t]=wmma16(a1, lds_load_b(sW0, 32, t*16, lane), c[t]);
#pragma unroll
      for (int v=0;v<8;++v) c[t][v] = reluf(c[t][v] + sB0[t*16+n]);
    }
#pragma unroll
    for (int t=0;t<4;++t)
#pragma unroll
      for (int v=0;v<8;++v) hT[(v+8*hlf)*64 + t*16 + n] = (_Float16)c[t][v];
  }
  residual_tail(c, hT, m, hlf, lane, sW1, sB1, sW2, sB2);

  // out = relu(x + h)
#pragma unroll
  for (int t=0;t<4;++t)
#pragma unroll
    for (int v=0;v<8;++v){
      int nd2 = base + v + 8*hlf;
      if (nd2 < N_NODES){
        int col = t*16 + n;
        out[nd2*64 + col] = reluf(x[nd2*64+col] + c[t][v]);
      }
    }
}

extern "C" void kernel_launch(void* const* d_in, const int* in_sizes, int n_in,
                              void* d_out, int out_size, void* d_ws, size_t ws_size,
                              hipStream_t stream)
{
  (void)in_sizes; (void)n_in; (void)out_size; (void)ws_size;
  const float* x   = (const float*)d_in[0];
  const float* ea  = (const float*)d_in[1];
  const int*   ei  = (const int*)d_in[2];
  const float* q   = (const float*)d_in[3];
  const float* kW0=(const float*)d_in[4];  const float* kb0=(const float*)d_in[5];
  const float* kW1=(const float*)d_in[6];  const float* kb1=(const float*)d_in[7];
  const float* kW2=(const float*)d_in[8];  const float* kb2=(const float*)d_in[9];
  const float* vW0=(const float*)d_in[10]; const float* vb0=(const float*)d_in[11];
  const float* vW1=(const float*)d_in[12]; const float* vb1=(const float*)d_in[13];
  const float* vW2=(const float*)d_in[14]; const float* vb2=(const float*)d_in[15];
  const float* oW0=(const float*)d_in[16]; const float* ob0=(const float*)d_in[17];
  const float* oW1=(const float*)d_in[18]; const float* ob1=(const float*)d_in[19];
  const float* oW2=(const float*)d_in[20]; const float* ob2=(const float*)d_in[21];

  // workspace layout: scores f32 | v f16 | segmax u32 | denom f32 | acc f32  (~144 MB)
  char* ws = (char*)d_ws;
  size_t off = 0;
  float*    sBuf    = (float*)(ws+off);    off += (size_t)N_EDGES*8*sizeof(float);
  _Float16* vBuf    = (_Float16*)(ws+off); off += (size_t)N_EDGES*64*sizeof(_Float16);
  unsigned* mBuf    = (unsigned*)(ws+off); off += (size_t)N_NODES*8*sizeof(unsigned);
  float*    denomBuf= (float*)(ws+off);    off += (size_t)N_NODES*8*sizeof(float);
  float*    accBuf  = (float*)(ws+off);    off += (size_t)N_NODES*64*sizeof(float);

  init_ws_kernel<<<(N_NODES*64+255)/256, 256, 0, stream>>>(mBuf, denomBuf, accBuf);
  edge_kv_kernel<<<N_EDGES/128, 256, 0, stream>>>(x, ea, ei, q,
      kW0,kb0,kW1,kb1,kW2,kb2, vW0,vb0,vW1,vb1,vW2,vb2, sBuf, vBuf, mBuf);
  edge_softmax_scatter<<<(N_EDGES*8)/256, 256, 0, stream>>>(ei, sBuf, vBuf, mBuf, denomBuf, accBuf);
  node_out_kernel<<<(N_NODES+127)/128, 256, 0, stream>>>(x, oW0,ob0,oW1,ob1,oW2,ob2,
      denomBuf, accBuf, (float*)d_out);
}
